// RetinaNet_10496900072156
// MI455X (gfx1250) — compile-verified, matrix operand order checked by reading, not run on hardware
//
#include <hip/hip_runtime.h>
#include <math.h>

// ---------------------------------------------------------------------------
// RetinaNet head for MI455X (gfx1250, wave32, WMMA + TDM).
// Convs as implicit GEMM: v_wmma_f32_16x16x32_f16, NHWC f16 activations,
// weights DMA'd to LDS via tensor_load_to_lds (double buffered, TENSORcnt).
// ---------------------------------------------------------------------------

typedef __attribute__((ext_vector_type(16))) _Float16 v16h;
typedef __attribute__((ext_vector_type(8)))  float    v8f;
typedef __attribute__((ext_vector_type(4)))  unsigned uint4v;
typedef __attribute__((ext_vector_type(8)))  unsigned uint8v;

union Frag { v16h v; uint4 q[2]; };

#define NN 2
#define CC 256
#define AA 9
#define KK 80
#define RR 120087
#define KTOT 2304           // 256 * 9
#define TOPK_N 1000
#define SCALE_CLAMP 4.135166556742356f   // log(1000/16)

static __device__ inline unsigned pack2h(float a, float b) {
    union { _Float16 h[2]; unsigned u; } v;
    v.h[0] = (_Float16)a; v.h[1] = (_Float16)b; return v.u;
}

// ---------------------------------------------------------------------------
// TDM: DMA a 2D tile (tile0 x tile1, element=2B) from a row-major tensor into
// LDS. dim0_rem/dim1_rem are remaining tensor extents from the tile origin:
// rows/cols beyond them are zero-filled by the TDM (OOB reads return zero).
// D# layout per CDNA5 ISA.
// ---------------------------------------------------------------------------
static __device__ inline void tdm_load_2d(const void* gaddr, unsigned lds_off,
                                          unsigned dim0_rem, unsigned dim1_rem,
                                          unsigned tile0, unsigned tile1,
                                          unsigned stride0) {
    unsigned long long ga = (unsigned long long)gaddr;
    uint4v g0;
    g0.x = 1u;                                   // count=1, user descriptor
    g0.y = lds_off;                              // lds_addr (bytes)
    g0.z = (unsigned)ga;                         // global_addr[31:0]
    g0.w = (unsigned)((ga >> 32) & 0x01FFFFFFu) | (2u << 30);  // [56:32]|type=2
    uint8v g1;
    g1.s0 = 0x00010000u;                         // data_size=2B, no pad/iter/mask
    g1.s1 = (dim0_rem & 0xFFFFu) << 16;          // tensor_dim0[15:0]
    g1.s2 = ((dim0_rem >> 16) & 0xFFFFu) | ((dim1_rem & 0xFFFFu) << 16);
    g1.s3 = ((dim1_rem >> 16) & 0xFFFFu) | ((tile0 & 0xFFFFu) << 16);
    g1.s4 = tile1 & 0xFFFFu;                     // tile_dim1 (tile_dim2 = 0)
    g1.s5 = stride0;                             // tensor_dim0_stride[31:0]
    g1.s6 = 0u;
    g1.s7 = 0u;
    asm volatile("tensor_load_to_lds %0, %1" :: "s"(g0), "s"(g1) : "memory");
}

// ---------------------------------------------------------------------------
// feat (N,256,H,W) f32  ->  X (N*H*W, 256) f16  (NHWC)
// ---------------------------------------------------------------------------
__global__ __launch_bounds__(256) void feat_to_nhwc(const float* __restrict__ f,
                                                    _Float16* __restrict__ X,
                                                    int H, int W) {
    int total = NN * CC * H * W;
    int i = blockIdx.x * 256 + threadIdx.x;
    if (i >= total) return;
    int c = i & 255;
    int s = i >> 8;
    int HW = H * W;
    int n = s / HW, p = s - n * HW;
    X[i] = (_Float16)f[(n * CC + c) * HW + p];
}

// ---------------------------------------------------------------------------
// w (Cout,256,3,3) f32 OIHW  ->  Wr (Cout, 2304) f16, k = (ky*3+kx)*256 + cin
// ---------------------------------------------------------------------------
__global__ __launch_bounds__(256) void reorder_w(const float* __restrict__ w,
                                                 _Float16* __restrict__ Wr,
                                                 int Cout) {
    int total = Cout * KTOT;
    int i = blockIdx.x * 256 + threadIdx.x;
    if (i >= total) return;
    int o = i / KTOT, k = i - o * KTOT;
    int tap = k >> 8, cin = k & 255;
    int ky = tap / 3, kx = tap - ky * 3;
    Wr[i] = (_Float16)w[((o * CC + cin) * 3 + ky) * 3 + kx];
}

// ---------------------------------------------------------------------------
// 3x3 SAME conv, implicit GEMM.
//   X  : NHWC f16 activations (N*H*W, 256)
//   Wr : (Cout, 2304) f16, K-contiguous rows
//   mode 0: Yh = relu(conv+bias) f16 NHWC  (Cout == 256)
//   mode 1: Yf[n*outStrideN + p*Cout + outBase + c] = conv+bias  (f32)
// Block 256 thr = 8 waves. Tile 128 cout x 64 spatial; wave = 32x32 (4 WMMA).
// A staged by TDM (double-buffered), B by per-thread b128 loads (im2col).
// ---------------------------------------------------------------------------
__global__ __launch_bounds__(256) void conv3x3_wmma(
        const _Float16* __restrict__ X, const _Float16* __restrict__ Wr,
        const float* __restrict__ bias,
        _Float16* __restrict__ Yh, float* __restrict__ Yf,
        int Cout, int H, int W, int mode, long outStrideN, long outBase) {
    __shared__ _Float16 Alds[2][128 * 32];   // ping-pong [m][k]
    __shared__ _Float16 Blds[64 * 32];       // [s][k]

    const int tid = threadIdx.x;
    const int HW = H * W, S = NN * HW;
    const int s0 = blockIdx.x * 64;
    const int m0 = blockIdx.y * 128;

    // per-thread B-tile position (one spatial row, 8 halves per chunk)
    const int sB = s0 + (tid >> 2);
    int nB = 0, yB = 0, xB = 0;
    const bool sBvalid = (sB < S);
    if (sBvalid) { nB = sB / HW; int p = sB - nB * HW; yB = p / W; xB = p - yB * W; }

    const int wv = tid >> 5, l = tid & 31;
    const int wvM = wv & 3, wvN = wv >> 2;       // 4 x 2 wave grid
    const int h = l >> 4, lr = l & 15;

    const _Float16* WrTile = Wr + (size_t)m0 * KTOT;
    const unsigned ldsA[2] = { (unsigned)(size_t)(void*)&Alds[0][0],
                               (unsigned)(size_t)(void*)&Alds[1][0] };
    const unsigned dim1rem = (unsigned)(Cout - m0);

    v8f acc00 = {}, acc01 = {}, acc10 = {}, acc11 = {};

    if (wv == 0)   // prime the pipeline: weight tile for k-step 0
        tdm_load_2d(WrTile, ldsA[0], KTOT, dim1rem, 32, 128, KTOT);

    for (int kk = 0; kk < KTOT; kk += 32) {
        const int it = kk >> 5;
        const int tap = kk >> 8;
        const int cin0 = kk & 255;
        // ---- stage B tile (64 x 32): one b128 per thread, SAME padding
        uint4 bv; bv.x = bv.y = bv.z = bv.w = 0u;
        const int yy = yB + tap / 3 - 1;
        const int xx = xB + (tap - (tap / 3) * 3) - 1;
        if (sBvalid && (unsigned)yy < (unsigned)H && (unsigned)xx < (unsigned)W) {
            const _Float16* src =
                X + (((long)nB * H + yy) * W + xx) * CC + cin0 + (tid & 3) * 8;
            bv = *reinterpret_cast<const uint4*>(src);
            __builtin_prefetch(src + CC, 0, 1);   // global_prefetch_b8
        }
        *reinterpret_cast<uint4*>(Blds + (tid >> 2) * 32 + (tid & 3) * 8) = bv;
        // ---- wave 0: DMA next weight tile, then wait for current one
        if (wv == 0) {
            if (kk + 32 < KTOT) {
                tdm_load_2d(WrTile + kk + 32, ldsA[(it + 1) & 1],
                            (unsigned)(KTOT - (kk + 32)), dim1rem, 32, 128, KTOT);
                __builtin_amdgcn_s_wait_tensorcnt(1);
            } else {
                __builtin_amdgcn_s_wait_tensorcnt(0);
            }
        }
        __syncthreads();
        // ---- fragments: per-lane K split {h*8.., 16+h*8..}
        const _Float16* Ab = &Alds[it & 1][0];
        Frag fa0, fa1, fb0, fb1;
        const uint4* Ar0 = reinterpret_cast<const uint4*>(Ab + (wvM * 32 + lr) * 32);
        const uint4* Ar1 = reinterpret_cast<const uint4*>(Ab + (wvM * 32 + 16 + lr) * 32);
        fa0.q[0] = Ar0[h]; fa0.q[1] = Ar0[2 + h];
        fa1.q[0] = Ar1[h]; fa1.q[1] = Ar1[2 + h];
        const uint4* Br0 = reinterpret_cast<const uint4*>(Blds + (wvN * 32 + lr) * 32);
        const uint4* Br1 = reinterpret_cast<const uint4*>(Blds + (wvN * 32 + 16 + lr) * 32);
        fb0.q[0] = Br0[h]; fb0.q[1] = Br0[2 + h];
        fb1.q[0] = Br1[h]; fb1.q[1] = Br1[2 + h];
        // ---- 4 WMMAs per wave per K-step (2x2 register blocking)
        acc00 = __builtin_amdgcn_wmma_f32_16x16x32_f16(false, fa0.v, false, fb0.v,
                                                       (short)0, acc00, false, false);
        acc01 = __builtin_amdgcn_wmma_f32_16x16x32_f16(false, fa0.v, false, fb1.v,
                                                       (short)0, acc01, false, false);
        acc10 = __builtin_amdgcn_wmma_f32_16x16x32_f16(false, fa1.v, false, fb0.v,
                                                       (short)0, acc10, false, false);
        acc11 = __builtin_amdgcn_wmma_f32_16x16x32_f16(false, fa1.v, false, fb1.v,
                                                       (short)0, acc11, false, false);
        __syncthreads();
    }

    // ---- epilogue: 4 tiles; lane column = lr, rows cBase..cBase+7 contiguous
    v8f accs[4] = { acc00, acc01, acc10, acc11 };
    #pragma unroll
    for (int mi = 0; mi < 2; ++mi) {
        const int cBase = m0 + wvM * 32 + mi * 16 + h * 8;
        if (cBase >= Cout) continue;
        #pragma unroll
        for (int ni = 0; ni < 2; ++ni) {
            const v8f a = accs[mi * 2 + ni];
            const int sOut = s0 + wvN * 32 + ni * 16 + lr;
            if (sOut >= S) continue;
            const int n = sOut / HW, p = sOut - n * HW;
            const long off = (long)n * outStrideN + (long)p * Cout + outBase + cBase;
            if (mode == 0) {          // f16 + ReLU (Cout == 256), vector store
                float v[8];
                #pragma unroll
                for (int vi = 0; vi < 8; ++vi)
                    v[vi] = fmaxf(a[vi] + bias[cBase + vi], 0.0f);
                uint4 pk;
                pk.x = pack2h(v[0], v[1]); pk.y = pack2h(v[2], v[3]);
                pk.z = pack2h(v[4], v[5]); pk.w = pack2h(v[6], v[7]);
                *reinterpret_cast<uint4*>(Yh + off) = pk;
            } else {                  // f32, masked (Cout 720 / 36)
                #pragma unroll
                for (int vi = 0; vi < 8; ++vi) {
                    int c = cBase + vi;
                    if (c < Cout) Yf[off + vi] = a[vi] + bias[c];
                }
            }
        }
    }
}

// ---------------------------------------------------------------------------
// focal loss: two-pass deterministic reduction
// ---------------------------------------------------------------------------
static __device__ inline float softplusf(float x) {
    return x > 0.f ? x + log1pf(expf(-x)) : log1pf(expf(x));
}

__global__ __launch_bounds__(256) void focal_part1(const float* __restrict__ logits,
                                                   const float* __restrict__ targets,
                                                   float* __restrict__ partials,
                                                   long total) {
    long i0 = (long)blockIdx.x * 256 * 32 + threadIdx.x;
    float sum = 0.f;
    #pragma unroll 4
    for (int j = 0; j < 32; ++j) {
        long i = i0 + (long)j * 256;
        if (i < total) {
            float x = logits[i], t = targets[i];
            float ce = t * softplusf(-x) + (1.f - t) * softplusf(x);
            float pr = 1.f / (1.f + expf(-x));
            float pt = pr * t + (1.f - pr) * (1.f - t);
            float om = 1.f - pt;
            float at = 0.25f * t + 0.75f * (1.f - t);
            sum += at * ce * om * om;
        }
    }
    __shared__ float red[256];
    red[threadIdx.x] = sum; __syncthreads();
    for (int s = 128; s > 0; s >>= 1) {
        if (threadIdx.x < s) red[threadIdx.x] += red[threadIdx.x + s];
        __syncthreads();
    }
    if (threadIdx.x == 0) partials[blockIdx.x] = red[0];
}

__global__ __launch_bounds__(256) void focal_part2(const float* __restrict__ partials,
                                                   int nb, float* __restrict__ out) {
    float s = 0.f;
    for (int i = threadIdx.x; i < nb; i += 256) s += partials[i];
    __shared__ float red[256];
    red[threadIdx.x] = s; __syncthreads();
    for (int k = 128; k > 0; k >>= 1) {
        if (threadIdx.x < k) red[threadIdx.x] += red[threadIdx.x + k];
        __syncthreads();
    }
    if (threadIdx.x == 0) out[0] = red[0] * 0.01f;   // / LOSS_NORMALIZER
}

// ---------------------------------------------------------------------------
// box transform
// ---------------------------------------------------------------------------
__global__ __launch_bounds__(256) void box_apply(const float* __restrict__ deltas,
                                                 const float* __restrict__ anchors,
                                                 float* __restrict__ out) {
    int i = blockIdx.x * 256 + threadIdx.x;       // over N*R
    if (i >= NN * RR) return;
    int r = i % RR;
    float x0 = anchors[r * 4], y0 = anchors[r * 4 + 1];
    float x1 = anchors[r * 4 + 2], y1 = anchors[r * 4 + 3];
    float w = x1 - x0, hgt = y1 - y0;
    float cx = x0 + 0.5f * w, cy = y0 + 0.5f * hgt;
    float dx = deltas[i * 4], dy = deltas[i * 4 + 1];
    float dw = fminf(deltas[i * 4 + 2], SCALE_CLAMP);
    float dh = fminf(deltas[i * 4 + 3], SCALE_CLAMP);
    float pcx = dx * w + cx, pcy = dy * hgt + cy;
    float pw = expf(dw) * w, ph = expf(dh) * hgt;
    out[i * 4 + 0] = pcx - 0.5f * pw;
    out[i * 4 + 1] = pcy - 0.5f * ph;
    out[i * 4 + 2] = pcx + 0.5f * pw;
    out[i * 4 + 3] = pcy + 0.5f * ph;
}

// ---------------------------------------------------------------------------
// top-k pipeline: 1024-bin histogram on monotone key of logit bits
// ---------------------------------------------------------------------------
static __device__ inline unsigned mapkey(float f) {
    unsigned u = __float_as_uint(f);
    return ((int)u >= 0) ? (u | 0x80000000u) : ~u;
}

__global__ __launch_bounds__(256) void fill_f32(float* p, float v, int n) {
    int i = blockIdx.x * 256 + threadIdx.x; if (i < n) p[i] = v;
}
__global__ __launch_bounds__(256) void zero_u32(unsigned* p, int n) {
    int i = blockIdx.x * 256 + threadIdx.x; if (i < n) p[i] = 0u;
}

__global__ __launch_bounds__(256) void topk_hist(const float* __restrict__ logits,
                                                 unsigned* __restrict__ hist) {
    const long RK = (long)RR * KK;
    int n = blockIdx.y;
    __shared__ unsigned hl[1024];
    for (int i = threadIdx.x; i < 1024; i += 256) hl[i] = 0u;
    __syncthreads();
    long e0 = (long)blockIdx.x * 4096 + threadIdx.x;
    for (int j = 0; j < 16; ++j) {
        long e = e0 + (long)j * 256;
        if (e < RK) atomicAdd(&hl[mapkey(logits[n * RK + e]) >> 22], 1u);
    }
    __syncthreads();
    for (int i = threadIdx.x; i < 1024; i += 256)
        if (hl[i]) atomicAdd(&hist[n * 1024 + i], hl[i]);
}

__global__ __launch_bounds__(1024) void topk_select(const unsigned* __restrict__ hist,
                                                    int* __restrict__ selT,
                                                    int* __restrict__ selAbove) {
    int n = blockIdx.x, t = threadIdx.x;
    __shared__ unsigned s[1024];
    s[t] = hist[n * 1024 + t];
    __syncthreads();
    for (int off = 1; off < 1024; off <<= 1) {   // inclusive suffix sum
        unsigned add = (t + off < 1024) ? s[t + off] : 0u;
        __syncthreads();
        s[t] += add;
        __syncthreads();
    }
    unsigned incl = s[t];
    unsigned above = (t < 1023) ? s[t + 1] : 0u;
    if (incl >= (unsigned)TOPK_N && above < (unsigned)TOPK_N) {
        selT[n] = t; selAbove[n] = (int)above;
    }
}

__global__ __launch_bounds__(256) void topk_compact(const float* __restrict__ logits,
                                                    const int* __restrict__ selT,
                                                    const int* __restrict__ selAbove,
                                                    int* __restrict__ cnt,
                                                    float* __restrict__ candVal,
                                                    int* __restrict__ candIdx) {
    const long RK = (long)RR * KK;
    int n = blockIdx.y;
    int T = selT[n], above = selAbove[n], need = TOPK_N - above;
    long e0 = (long)blockIdx.x * 4096 + threadIdx.x;
    for (int j = 0; j < 16; ++j) {
        long e = e0 + (long)j * 256;
        if (e >= RK) break;
        float x = logits[n * RK + e];
        int b = (int)(mapkey(x) >> 22);
        if (b > T) {
            int pos = atomicAdd(&cnt[2 * n], 1);
            candVal[n * 1024 + pos] = x; candIdx[n * 1024 + pos] = (int)e;
        } else if (b == T) {
            int pos = atomicAdd(&cnt[2 * n + 1], 1);
            if (pos < need) {
                int sl = above + pos;
                candVal[n * 1024 + sl] = x; candIdx[n * 1024 + sl] = (int)e;
            }
        }
    }
}

__global__ __launch_bounds__(1024) void topk_sort(const float* __restrict__ candVal,
                                                  const int* __restrict__ candIdx,
                                                  float* __restrict__ outS,
                                                  float* __restrict__ outI) {
    int n = blockIdx.x, t = threadIdx.x;
    __shared__ float kv[1024];
    __shared__ int   ki[1024];
    kv[t] = candVal[n * 1024 + t];
    ki[t] = candIdx[n * 1024 + t];
    __syncthreads();
    for (int k2 = 2; k2 <= 1024; k2 <<= 1)
        for (int j = k2 >> 1; j > 0; j >>= 1) {
            int ixj = t ^ j;
            if (ixj > t) {
                bool desc = ((t & k2) == 0);
                float a = kv[t], b = kv[ixj];
                bool sw = desc ? (a < b) : (a > b);
                if (sw) {
                    kv[t] = b; kv[ixj] = a;
                    int ia = ki[t]; ki[t] = ki[ixj]; ki[ixj] = ia;
                }
            }
            __syncthreads();
        }
    if (t < TOPK_N) {
        float x = kv[t];
        outS[n * TOPK_N + t] = 1.f / (1.f + expf(-x));
        ((int*)outI)[n * TOPK_N + t] = ki[t];
    }
}

// ---------------------------------------------------------------------------
// launcher
// ---------------------------------------------------------------------------
extern "C" void kernel_launch(void* const* d_in, const int* in_sizes, int n_in,
                              void* d_out, int out_size, void* d_ws, size_t ws_size,
                              hipStream_t stream) {
    (void)in_sizes; (void)n_in; (void)out_size; (void)ws_size;
    static const int HWs[5]    = {100, 50, 25, 13, 7};
    static const int rowoff[5] = {0, 90000, 112500, 118125, 119646};

    const float* feat[5];
    for (int i = 0; i < 5; ++i) feat[i] = (const float*)d_in[i];
    const float* cls_w        = (const float*)d_in[5];
    const float* cls_b        = (const float*)d_in[6];
    const float* bbox_w       = (const float*)d_in[7];
    const float* bbox_b       = (const float*)d_in[8];
    const float* cls_score_w  = (const float*)d_in[9];
    const float* cls_score_b  = (const float*)d_in[10];
    const float* bbox_pred_w  = (const float*)d_in[11];
    const float* bbox_pred_b  = (const float*)d_in[12];
    const float* anchors      = (const float*)d_in[13];
    const float* targets      = (const float*)d_in[14];

    // ---- workspace carving
    char* p = (char*)d_ws;
    auto alloc = [&](size_t bytes) -> char* {
        char* r = p; p += (bytes + 255) & ~(size_t)255; return r;
    };
    const size_t SMAX = 2 * 100 * 100;                       // 20000
    _Float16* X0 = (_Float16*)alloc(SMAX * CC * sizeof(_Float16));
    _Float16* A1 = (_Float16*)alloc(SMAX * CC * sizeof(_Float16));
    _Float16* A2 = (_Float16*)alloc(SMAX * CC * sizeof(_Float16));
    _Float16* WrCls   = (_Float16*)alloc((size_t)4 * CC * KTOT * sizeof(_Float16));
    _Float16* WrBox   = (_Float16*)alloc((size_t)4 * CC * KTOT * sizeof(_Float16));
    _Float16* WrScore = (_Float16*)alloc((size_t)(AA * KK) * KTOT * sizeof(_Float16));
    _Float16* WrPred  = (_Float16*)alloc((size_t)(AA * 4) * KTOT * sizeof(_Float16));
    float*    logits  = (float*)alloc((size_t)NN * RR * KK * sizeof(float));
    float*    deltas  = (float*)alloc((size_t)NN * RR * 4 * sizeof(float));
    float*    partials = (float*)alloc(4096 * sizeof(float));
    unsigned* hist    = (unsigned*)alloc((size_t)NN * 1024 * sizeof(unsigned));
    int*      selT    = (int*)alloc(NN * sizeof(int));
    int*      selAb   = (int*)alloc(NN * sizeof(int));
    int*      cnt     = (int*)alloc(2 * NN * sizeof(int));
    float*    candVal = (float*)alloc((size_t)NN * 1024 * sizeof(float));
    int*      candIdx = (int*)alloc((size_t)NN * 1024 * sizeof(int));

    float* out       = (float*)d_out;
    float* outBoxes  = out + 1;
    float* outScores = out + 1 + (size_t)NN * RR * 4;
    float* outIdx    = outScores + (size_t)NN * TOPK_N;

    // ---- weight reorder (OIHW f32 -> [Cout][2304] f16)
    {
        int tot = CC * KTOT, nb = (tot + 255) / 256;
        for (int i = 0; i < 4; ++i) {
            reorder_w<<<nb, 256, 0, stream>>>(cls_w  + (size_t)i * CC * CC * 9,
                                              WrCls  + (size_t)i * CC * KTOT, CC);
            reorder_w<<<nb, 256, 0, stream>>>(bbox_w + (size_t)i * CC * CC * 9,
                                              WrBox  + (size_t)i * CC * KTOT, CC);
        }
        int totS = AA * KK * KTOT;
        reorder_w<<<(totS + 255) / 256, 256, 0, stream>>>(cls_score_w, WrScore, AA * KK);
        int totP = AA * 4 * KTOT;
        reorder_w<<<(totP + 255) / 256, 256, 0, stream>>>(bbox_pred_w, WrPred, AA * 4);
    }

    // ---- per level: convert + head chains + predictors (WMMA conv)
    for (int lvl = 0; lvl < 5; ++lvl) {
        int H = HWs[lvl], W = HWs[lvl], HW = H * W, S = NN * HW;
        feat_to_nhwc<<<(S * CC + 255) / 256, 256, 0, stream>>>(feat[lvl], X0, H, W);

        dim3 gHead((S + 63) / 64, CC / 128);
        long sN = (long)HW * CC;
        // cls tower
        conv3x3_wmma<<<gHead, 256, 0, stream>>>(X0, WrCls + 0 * (size_t)CC * KTOT,
                                                cls_b + 0 * CC, A1, nullptr, CC, H, W, 0, sN, 0);
        conv3x3_wmma<<<gHead, 256, 0, stream>>>(A1, WrCls + 1 * (size_t)CC * KTOT,
                                                cls_b + 1 * CC, A2, nullptr, CC, H, W, 0, sN, 0);
        conv3x3_wmma<<<gHead, 256, 0, stream>>>(A2, WrCls + 2 * (size_t)CC * KTOT,
                                                cls_b + 2 * CC, A1, nullptr, CC, H, W, 0, sN, 0);
        conv3x3_wmma<<<gHead, 256, 0, stream>>>(A1, WrCls + 3 * (size_t)CC * KTOT,
                                                cls_b + 3 * CC, A2, nullptr, CC, H, W, 0, sN, 0);
        dim3 gScore((S + 63) / 64, (AA * KK + 127) / 128);
        conv3x3_wmma<<<gScore, 256, 0, stream>>>(A2, WrScore, cls_score_b, nullptr, logits,
                                                 AA * KK, H, W, 1,
                                                 (long)RR * KK, (long)rowoff[lvl] * KK);
        // bbox tower
        conv3x3_wmma<<<gHead, 256, 0, stream>>>(X0, WrBox + 0 * (size_t)CC * KTOT,
                                                bbox_b + 0 * CC, A1, nullptr, CC, H, W, 0, sN, 0);
        conv3x3_wmma<<<gHead, 256, 0, stream>>>(A1, WrBox + 1 * (size_t)CC * KTOT,
                                                bbox_b + 1 * CC, A2, nullptr, CC, H, W, 0, sN, 0);
        conv3x3_wmma<<<gHead, 256, 0, stream>>>(A2, WrBox + 2 * (size_t)CC * KTOT,
                                                bbox_b + 2 * CC, A1, nullptr, CC, H, W, 0, sN, 0);
        conv3x3_wmma<<<gHead, 256, 0, stream>>>(A1, WrBox + 3 * (size_t)CC * KTOT,
                                                bbox_b + 3 * CC, A2, nullptr, CC, H, W, 0, sN, 0);
        dim3 gPred((S + 63) / 64, 1);
        conv3x3_wmma<<<gPred, 256, 0, stream>>>(A2, WrPred, bbox_pred_b, nullptr, deltas,
                                                AA * 4, H, W, 1,
                                                (long)RR * 4, (long)rowoff[lvl] * 4);
    }

    // ---- focal loss (deterministic 2-pass)
    const long totalLog = (long)NN * RR * KK;
    const int nb1 = (int)((totalLog + 8191) / 8192);
    focal_part1<<<nb1, 256, 0, stream>>>(logits, targets, partials, totalLog);
    focal_part2<<<1, 256, 0, stream>>>(partials, nb1, out);

    // ---- box transform
    box_apply<<<(NN * RR + 255) / 256, 256, 0, stream>>>(deltas, anchors, outBoxes);

    // ---- top-k per image over R*K
    const long RKl = (long)RR * KK;
    zero_u32<<<(NN * 1024 + 255) / 256, 256, 0, stream>>>(hist, NN * 1024);
    dim3 gH((unsigned)((RKl + 4095) / 4096), NN);
    topk_hist<<<gH, 256, 0, stream>>>(logits, hist);
    topk_select<<<NN, 1024, 0, stream>>>(hist, selT, selAb);
    zero_u32<<<1, 256, 0, stream>>>((unsigned*)cnt, 2 * NN);
    fill_f32<<<(NN * 1024 + 255) / 256, 256, 0, stream>>>(candVal, -INFINITY, NN * 1024);
    topk_compact<<<gH, 256, 0, stream>>>(logits, selT, selAb, cnt, candVal, candIdx);
    topk_sort<<<NN, 1024, 0, stream>>>(candVal, candIdx, outScores, outIdx);
}
